// HEARTForSequenceClassification_13469017440330
// MI455X (gfx1250) — matrix operand order repaired and checked by reading, not compiled, not run on hardware
//
#include <hip/hip_runtime.h>

#define B_  4
#define L_  512
#define D_  768
#define H_  12
#define DK_ 64
#define E_  64
#define R_  32
#define T_  8

typedef __bf16 bf16_t;
typedef __attribute__((ext_vector_type(16))) __bf16 v16bf;
typedef __attribute__((ext_vector_type(8)))  float  v8f;

__device__ __forceinline__ bf16_t f2b(float f) {
  unsigned u = __builtin_bit_cast(unsigned, f);
  u += 0x7FFFu + ((u >> 16) & 1u);           // round-to-nearest-even
  unsigned short h = (unsigned short)(u >> 16);
  return __builtin_bit_cast(bf16_t, h);
}

union FragU { v16bf v; float4 q[2]; };

// A fragment: 16x32 bf16 tile, row-major in LDS with row stride `ld` elements.
// Lane l: m = l&15, half = l>>4. elems 0..7 -> K = k0+half*8..+7 ; elems 8..15 -> K = k0+16+half*8..+7
__device__ __forceinline__ v16bf frag_a(const bf16_t* base, int ld, int k0) {
  int l = threadIdx.x & 31;
  int m = l & 15, hf = l >> 4;
  const bf16_t* p = base + m * ld + k0 + hf * 8;
  FragU f;
  f.q[0] = *(const float4*)(p);
  f.q[1] = *(const float4*)(p + 16);
  return f.v;
}

// B fragment: K x 16 matrix stored TRANSPOSED in LDS as N x K (row n, stride ld).
// Lane l: n = l&15, half = l>>4. elems 0..7 -> K = k0+half*8..+7 ; elems 8..15 -> K = k0+16+half*8..+7
__device__ __forceinline__ v16bf frag_b(const bf16_t* base, int ld, int k0) {
  int l = threadIdx.x & 31;
  int n = l & 15, hf = l >> 4;
  const bf16_t* p = base + n * ld + k0 + hf * 8;
  FragU f;
  f.q[0] = *(const float4*)(p);
  f.q[1] = *(const float4*)(p + 16);
  return f.v;
}

__device__ __forceinline__ v8f wmma_bf16(v16bf a, v16bf b, v8f c) {
  return __builtin_amdgcn_wmma_f32_16x16x32_bf16(false, a, false, b, (short)0, c, false, false);
}

// CDNA5 async global->LDS (tracked by ASYNCcnt, bypasses VGPRs).
// lds_off = group-segment byte offset (low 32 bits of a generic LDS pointer).
__device__ __forceinline__ void async_ld_b128(const void* lds_ptr, const void* gptr) {
  unsigned lds_off = (unsigned)(unsigned long long)lds_ptr;
  asm volatile("global_load_async_to_lds_b128 %0, %1, off"
               :: "v"(lds_off), "v"(gptr) : "memory");
}
__device__ __forceinline__ void wait_async() {
  asm volatile("s_wait_asynccnt 0x0" ::: "memory");
}

// ---------------------------------------------------------------- f32 -> bf16
__global__ void f2bf_kernel(const float* __restrict__ in, bf16_t* __restrict__ out, int n) {
  int i = blockIdx.x * 256 + threadIdx.x;
  if (i < n) out[i] = f2b(in[i]);
}

// ---------------------------------------------------------------- LayerNorm (f32 in, bf16 out)
__global__ void ln_kernel(const float* __restrict__ x, const float* __restrict__ g,
                          const float* __restrict__ bta, bf16_t* __restrict__ out, int Dn) {
  __shared__ float r1[256], r2[256];
  int row = blockIdx.x, t = threadIdx.x;
  float s1 = 0.f, s2 = 0.f;
  for (int d = t; d < Dn; d += 256) { float v = x[(size_t)row * Dn + d]; s1 += v; s2 += v * v; }
  r1[t] = s1; r2[t] = s2; __syncthreads();
  for (int s = 128; s > 0; s >>= 1) {
    if (t < s) { r1[t] += r1[t + s]; r2[t] += r2[t + s]; }
    __syncthreads();
  }
  float mean = r1[0] / Dn;
  float var  = r2[0] / Dn - mean * mean;
  float rstd = rsqrtf(var + 1e-5f);
  for (int d = t; d < Dn; d += 256) {
    float v = (x[(size_t)row * Dn + d] - mean) * rstd * g[d] + bta[d];
    out[(size_t)row * Dn + d] = f2b(v);
  }
}

// ---------------------------------------------------------------- Edge low-rank left/right
__global__ void lr_kernel(const float* __restrict__ x, const int* __restrict__ ttp,
                          const float* __restrict__ lt_W, const float* __restrict__ rt_W,
                          const float* __restrict__ ol_W, const float* __restrict__ ol_b,
                          const float* __restrict__ or_W,
                          float* __restrict__ leftout, float* __restrict__ rightout) {
  __shared__ float xs[D_];
  __shared__ float lbuf[R_], rbuf[R_];
  int blk = blockIdx.x, t = threadIdx.x;
  int tt = ttp[blk]; tt = tt < 0 ? 0 : (tt > T_ - 1 ? T_ - 1 : tt);
  for (int d = t; d < D_; d += 256) xs[d] = x[(size_t)blk * D_ + d];
  __syncthreads();
  int wave = t >> 5, lane = t & 31;
  for (int rr = 0; rr < 4; ++rr) {
    int r = wave + rr * 8;
    const float* lp = &lt_W[((size_t)tt * R_ + r) * D_];
    const float* rp = &rt_W[((size_t)tt * R_ + r) * D_];
    float a = 0.f, c = 0.f;
    for (int d = lane; d < D_; d += 32) { a += xs[d] * lp[d]; c += xs[d] * rp[d]; }
    for (int off = 16; off > 0; off >>= 1) { a += __shfl_xor(a, off, 32); c += __shfl_xor(c, off, 32); }
    if (lane == 0) { lbuf[r] = a; rbuf[r] = c; }
  }
  __syncthreads();
  if (t < E_) {
    float la = ol_b[t], ra = 0.f;
    for (int r = 0; r < R_; ++r) { la += lbuf[r] * ol_W[t * R_ + r]; ra += rbuf[r] * or_W[t * R_ + r]; }
    leftout[(size_t)blk * E_ + t]  = la;
    rightout[(size_t)blk * E_ + t] = ra;
  }
}

// ---------------------------------------------------------------- Fused edge: edge -> LN -> kse (WMMA) + eb (softcap)
__global__ void edge_kernel(const float* __restrict__ leftout, const float* __restrict__ rightout,
                            const float* __restrict__ wke_W, const float* __restrict__ wke_b,
                            const float* __restrict__ we_W, const float* __restrict__ we_b,
                            const float* __restrict__ ne_g, const float* __restrict__ ne_b,
                            bf16_t* __restrict__ kse, float* __restrict__ eb) {
  __shared__ __align__(16) bf16_t ne_s[64 * 64];
  __shared__ __align__(16) bf16_t wke_s[64 * 64];
  __shared__ float lo_s[64], g_s[64], bt_s[64], we_s[64];
  int bi = blockIdx.x;           // b*L + i
  int b  = bi / L_;
  int j0 = blockIdx.y * 64;
  int t  = threadIdx.x;
  if (t < 64) { lo_s[t] = leftout[(size_t)bi * E_ + t]; g_s[t] = ne_g[t]; bt_s[t] = ne_b[t]; we_s[t] = we_W[t]; }
  for (int i = t; i < 64 * 64; i += 128) wke_s[i] = f2b(wke_W[i]);
  __syncthreads();

  int r = t >> 1, part = t & 1;
  int j = j0 + r;
  const float* rp = &rightout[((size_t)(b * L_ + j)) * E_ + part * 32];
  float vals[32]; float s1 = 0.f, s2 = 0.f;
  for (int e = 0; e < 32; ++e) { float v = lo_s[part * 32 + e] + rp[e]; vals[e] = v; s1 += v; s2 += v * v; }
  s1 += __shfl_xor(s1, 1, 32); s2 += __shfl_xor(s2, 1, 32);
  float mean = s1 * (1.f / 64.f);
  float var  = s2 * (1.f / 64.f) - mean * mean;
  float rstd = rsqrtf(var + 1e-5f);
  float ep = 0.f;
  for (int e = 0; e < 32; ++e) {
    int ee = part * 32 + e;
    float nv = (vals[e] - mean) * rstd * g_s[ee] + bt_s[ee];
    ne_s[r * 64 + ee] = f2b(nv);
    ep += nv * we_s[ee];
  }
  ep += __shfl_xor(ep, 1, 32);
  if (part == 0) {
    float v = (ep + we_b[0]) * 0.70710678118654752f;  // * 2^-0.5
    v = 5.0f * tanhf(v * 0.2f);                        // softcap CAP=5
    eb[(size_t)bi * L_ + j] = v;
  }
  __syncthreads();

  int wave = t >> 5;
  int m0 = (wave >> 1) * 32, n0 = (wave & 1) * 32;
  v8f acc[2][2] = {};
  for (int k0 = 0; k0 < 64; k0 += 32) {
    v16bf a0 = frag_a(&ne_s[m0 * 64], 64, k0);
    v16bf a1 = frag_a(&ne_s[(m0 + 16) * 64], 64, k0);
    v16bf b0 = frag_b(&wke_s[n0 * 64], 64, k0);
    v16bf b1 = frag_b(&wke_s[(n0 + 16) * 64], 64, k0);
    acc[0][0] = wmma_bf16(a0, b0, acc[0][0]);
    acc[0][1] = wmma_bf16(a0, b1, acc[0][1]);
    acc[1][0] = wmma_bf16(a1, b0, acc[1][0]);
    acc[1][1] = wmma_bf16(a1, b1, acc[1][1]);
  }
  int l = t & 31, ln_ = l & 15, hf = l >> 4;
  for (int i = 0; i < 2; ++i)
    for (int jj = 0; jj < 2; ++jj) {
      int col = n0 + jj * 16 + ln_;
      float bv = wke_b[col];
      for (int rr = 0; rr < 8; ++rr) {
        int rowj = m0 + i * 16 + rr + 8 * hf;
        kse[((size_t)bi * L_ + (j0 + rowj)) * E_ + col] = f2b(acc[i][jj][rr] + bv);
      }
    }
}

// ---------------------------------------------------------------- Attention: QK^T + eb + mask -> softmax -> attn (bf16) -> ctx
__global__ void attn_kernel(const bf16_t* __restrict__ qb, const bf16_t* __restrict__ kb,
                            const bf16_t* __restrict__ vb, const float* __restrict__ eb,
                            const unsigned char* __restrict__ maskp,
                            bf16_t* __restrict__ attn_g, bf16_t* __restrict__ cat) {
  __shared__ __align__(16) bf16_t Aq[16 * 64];
  __shared__ __align__(16) bf16_t Ks[64 * 64];      // reused as Vt[dk][m]
  __shared__ __align__(16) float  Sc[16 * 512];
  __shared__ __align__(16) bf16_t At[16 * 512];
  __shared__ float  red[16 * 8];
  __shared__ float  rowmax[16], rowsum[16];
  int bh = blockIdx.x, b = bh / H_, h = bh % H_;
  int n0 = blockIdx.y * 16;
  int t = threadIdx.x, wave = t >> 5, l = t & 31, ln_ = l & 15, hf = l >> 4;

  for (int i = t; i < 16 * 64; i += 128) {
    int rr = i >> 6, c = i & 63;
    Aq[i] = qb[((size_t)(b * L_ + n0 + rr)) * D_ + h * DK_ + c];
  }
  // scores
  for (int ch = 0; ch < 8; ++ch) {
    __syncthreads();
    // async copy K tile straight into LDS (64x64 bf16, 512 B128 transfers)
    for (int i = t; i < 512; i += 128) {
      int rr = i >> 3, c = (i & 7) * 8;
      async_ld_b128(&Ks[rr * 64 + c],
                    &kb[((size_t)(b * L_ + ch * 64 + rr)) * D_ + h * DK_ + c]);
    }
    wait_async();
    __syncthreads();
    v8f acc = {};
    for (int k0 = 0; k0 < 64; k0 += 32) {
      v16bf a  = frag_a(Aq, 64, k0);
      v16bf bb = frag_b(&Ks[(wave * 16) * 64], 64, k0);
      acc = wmma_bf16(a, bb, acc);
    }
    int col = ch * 64 + wave * 16 + ln_;
    for (int rr = 0; rr < 8; ++rr) {
      int row = rr + 8 * hf;
      size_t mi = (size_t)(b * L_ + n0 + row) * L_ + col;
      float s = acc[rr] * 0.08838834764831845f + eb[mi];  // (2*DK)^-0.5
      if (maskp[mi]) s = -1e9f;
      Sc[row * 512 + col] = s;
    }
  }
  __syncthreads();
  // softmax over 512 per row (8 segs of 64)
  {
    int row = t >> 3, seg = t & 7;
    float mx = -1e30f;
    for (int c = seg * 64; c < seg * 64 + 64; ++c) mx = fmaxf(mx, Sc[row * 512 + c]);
    red[row * 8 + seg] = mx; __syncthreads();
    if (t < 16) { float m = -1e30f; for (int s = 0; s < 8; ++s) m = fmaxf(m, red[t * 8 + s]); rowmax[t] = m; }
    __syncthreads();
    float sm = 0.f;
    for (int c = seg * 64; c < seg * 64 + 64; ++c) {
      float e = __expf(Sc[row * 512 + c] - rowmax[row]);
      Sc[row * 512 + c] = e; sm += e;
    }
    red[row * 8 + seg] = sm; __syncthreads();
    if (t < 16) { float s0 = 0.f; for (int s = 0; s < 8; ++s) s0 += red[t * 8 + s]; rowsum[t] = s0; }
    __syncthreads();
    float inv = 1.f / rowsum[row];
    for (int c = seg * 64; c < seg * 64 + 64; ++c) {
      bf16_t ab = f2b(Sc[row * 512 + c] * inv);
      At[row * 512 + c] = ab;
      attn_g[((size_t)bh * L_ + n0 + row) * L_ + c] = ab;
    }
  }
  // ctx = attn @ V   (wave w owns dk columns w*16..+15)
  v8f accc = {};
  for (int ch = 0; ch < 8; ++ch) {
    __syncthreads();
    for (int i = t; i < 64 * 64; i += 128) {
      int rr = i >> 6, c = i & 63;  // rr = m in chunk, c = dk
      Ks[c * 64 + rr] = vb[((size_t)(b * L_ + ch * 64 + rr)) * D_ + h * DK_ + c];  // Vt[dk][m]
    }
    __syncthreads();
    for (int k0 = 0; k0 < 64; k0 += 32) {
      v16bf a  = frag_a(&At[ch * 64], 512, k0);
      v16bf bb = frag_b(&Ks[(wave * 16) * 64], 64, k0);
      accc = wmma_bf16(a, bb, accc);
    }
  }
  for (int rr = 0; rr < 8; ++rr) {
    int row = rr + 8 * hf;
    cat[((size_t)(b * L_ + n0 + row)) * (2 * D_) + h * DK_ + wave * 16 + ln_] = f2b(accc[rr]);
  }
}

// ---------------------------------------------------------------- ectx[b,:,n,:] = attn(H x L) @ kse[b,n] (L x E), heads padded to 16
__global__ void ectx_kernel(const bf16_t* __restrict__ attn_g, const bf16_t* __restrict__ kse,
                            bf16_t* __restrict__ ectx) {
  __shared__ __align__(16) bf16_t As[16 * 128];
  __shared__ __align__(16) bf16_t Kt[64 * 128];
  int bn = blockIdx.x, b = bn / L_, n = bn % L_;
  int t = threadIdx.x, wave = t >> 5, l = t & 31, ln_ = l & 15, hf = l >> 4;
  v8f acc = {};
  for (int mc = 0; mc < 4; ++mc) {
    __syncthreads();
    for (int i = t; i < 16 * 128; i += 128) {
      int hd = i >> 7, m = i & 127;
      As[i] = (hd < H_) ? attn_g[((size_t)(b * H_ + hd) * L_ + n) * L_ + mc * 128 + m] : f2b(0.f);
    }
    for (int i = t; i < 64 * 128; i += 128) {
      int e = i >> 7, m = i & 127;
      Kt[i] = kse[((size_t)bn * L_ + mc * 128 + m) * E_ + e];   // transpose to N x K
    }
    __syncthreads();
    for (int k0 = 0; k0 < 128; k0 += 32) {
      v16bf a  = frag_a(As, 128, k0);
      v16bf bb = frag_b(&Kt[(wave * 16) * 128], 128, k0);
      acc = wmma_bf16(a, bb, acc);
    }
  }
  for (int rr = 0; rr < 8; ++rr) {
    int hd = rr + 8 * hf;
    if (hd < H_)
      ectx[(size_t)bn * (H_ * E_) + hd * E_ + wave * 16 + ln_] = f2b(acc[rr]);
  }
}

// ---------------------------------------------------------------- Generic bf16 GEMM: C = A(MxK) * W(NxK)^T [+bias][gelu|residual]
// Double-buffered async global->LDS staging; mode 0: +bias -> bf16 ; 1: +bias,GELU -> bf16 ; 2: +bias +res(f32) -> f32
__global__ void gemm_kernel(const bf16_t* __restrict__ A, const bf16_t* __restrict__ W,
                            const float* __restrict__ bias, void* __restrict__ outp,
                            const float* __restrict__ res,
                            int M, int N, int K, int ldo, int col_off, int mode) {
  __shared__ __align__(16) bf16_t As[2][64 * 32];
  __shared__ __align__(16) bf16_t Bs[2][64 * 32];
  int t = threadIdx.x, wave = t >> 5, l = t & 31, ln_ = l & 15, hf = l >> 4;
  int mb = blockIdx.x * 64, nb = blockIdx.y * 64;
  int m0 = (wave >> 1) * 32, n0 = (wave & 1) * 32;
  v8f acc[2][2] = {};

  // each thread issues 2+2 async B128 transfers per tile
  int r_ = t >> 2, c_ = (t & 3) * 8;           // t in [0,128): rows 0..31, plus +32 below
  auto issue = [&](int buf, int k0) {
    async_ld_b128(&As[buf][r_ * 32 + c_],        &A[(size_t)(mb + r_) * K + k0 + c_]);
    async_ld_b128(&As[buf][(r_ + 32) * 32 + c_], &A[(size_t)(mb + r_ + 32) * K + k0 + c_]);
    async_ld_b128(&Bs[buf][r_ * 32 + c_],        &W[(size_t)(nb + r_) * K + k0 + c_]);
    async_ld_b128(&Bs[buf][(r_ + 32) * 32 + c_], &W[(size_t)(nb + r_ + 32) * K + k0 + c_]);
    if (k0 + 64 < K) {   // warm GL2 for the tile after next
      __builtin_prefetch(&A[(size_t)(mb + r_) * K + k0 + 32 + c_], 0, 1);
      __builtin_prefetch(&W[(size_t)(nb + r_) * K + k0 + 32 + c_], 0, 1);
    }
  };

  issue(0, 0);
  int buf = 0;
  for (int k0 = 0; k0 < K; k0 += 32) {
    wait_async();
    __syncthreads();           // all async writes into As/Bs[buf] visible; prior reads of buf^1 retired
    if (k0 + 32 < K) issue(buf ^ 1, k0 + 32);
    v16bf a0 = frag_a(&As[buf][m0 * 32], 32, 0);
    v16bf a1 = frag_a(&As[buf][(m0 + 16) * 32], 32, 0);
    v16bf b0 = frag_b(&Bs[buf][n0 * 32], 32, 0);
    v16bf b1 = frag_b(&Bs[buf][(n0 + 16) * 32], 32, 0);
    acc[0][0] = wmma_bf16(a0, b0, acc[0][0]);
    acc[0][1] = wmma_bf16(a0, b1, acc[0][1]);
    acc[1][0] = wmma_bf16(a1, b0, acc[1][0]);
    acc[1][1] = wmma_bf16(a1, b1, acc[1][1]);
    buf ^= 1;
  }
  for (int i = 0; i < 2; ++i)
    for (int j = 0; j < 2; ++j) {
      int col = nb + n0 + j * 16 + ln_;
      float bv = bias ? bias[col] : 0.f;
      for (int rr = 0; rr < 8; ++rr) {
        int row = mb + m0 + i * 16 + rr + 8 * hf;
        float v = acc[i][j][rr] + bv;
        if (mode == 1) v = 0.5f * v * (1.f + erff(v * 0.70710678118654752f));
        size_t oi = (size_t)row * ldo + col_off + col;
        if (mode == 2) ((float*)outp)[oi] = v + res[oi];
        else           ((bf16_t*)outp)[oi] = f2b(v);
      }
    }
}

// ================================================================ host
extern "C" void kernel_launch(void* const* d_in, const int* in_sizes, int n_in,
                              void* d_out, int out_size, void* d_ws, size_t ws_size,
                              hipStream_t stream) {
  const float* x      = (const float*)d_in[0];
  const int*   ttok   = (const int*)d_in[1];
  const unsigned char* mask = (const unsigned char*)d_in[2];
  const float* lt_W = (const float*)d_in[3];
  const float* rt_W = (const float*)d_in[4];
  const float* ol_W = (const float*)d_in[5];
  const float* ol_b = (const float*)d_in[6];
  const float* or_W = (const float*)d_in[7];
  const float* wq_W = (const float*)d_in[8];
  const float* wq_b = (const float*)d_in[9];
  const float* wk_W = (const float*)d_in[10];
  const float* wk_b = (const float*)d_in[11];
  const float* wv_W = (const float*)d_in[12];
  const float* wv_b = (const float*)d_in[13];
  const float* wke_W = (const float*)d_in[14];
  const float* wke_b = (const float*)d_in[15];
  const float* we_W  = (const float*)d_in[16];
  const float* we_b  = (const float*)d_in[17];
  const float* weo_W = (const float*)d_in[18];
  const float* weo_b = (const float*)d_in[19];
  const float* wo_W  = (const float*)d_in[20];
  const float* wo_b  = (const float*)d_in[21];
  const float* f1_W  = (const float*)d_in[22];
  const float* f1_b  = (const float*)d_in[23];
  const float* f2_W  = (const float*)d_in[24];
  const float* f2_b  = (const float*)d_in[25];
  const float* na_g = (const float*)d_in[26];
  const float* na_b = (const float*)d_in[27];
  const float* ne_g = (const float*)d_in[28];
  const float* ne_b = (const float*)d_in[29];
  const float* nf_g = (const float*)d_in[30];
  const float* nf_b = (const float*)d_in[31];

  const size_t BL = (size_t)B_ * L_;
  char* w = (char*)d_ws; size_t off = 0;
  auto alloc = [&](size_t bytes) -> void* {
    void* p = w + off; off = (off + bytes + 255) & ~(size_t)255; return p;
  };
  bf16_t* nx    = (bf16_t*)alloc(BL * D_ * 2);
  float*  lout  = (float*)alloc(BL * E_ * 4);
  float*  rout  = (float*)alloc(BL * E_ * 4);
  bf16_t* qb    = (bf16_t*)alloc(BL * D_ * 2);
  bf16_t* kb    = (bf16_t*)alloc(BL * D_ * 2);
  bf16_t* vb    = (bf16_t*)alloc(BL * D_ * 2);
  bf16_t* kse   = (bf16_t*)alloc(BL * L_ * E_ * 2);        // 134 MB
  float*  eb    = (float*)alloc(BL * L_ * 4);              // 4 MB
  bf16_t* attn  = (bf16_t*)alloc((size_t)B_ * H_ * L_ * L_ * 2);  // 25 MB
  bf16_t* ectx  = (bf16_t*)alloc(BL * (H_ * E_) * 2);
  bf16_t* cat   = (bf16_t*)alloc(BL * 2 * D_ * 2);
  float*  x1    = (float*)alloc(BL * D_ * 4);
  bf16_t* nx2   = (bf16_t*)alloc(BL * D_ * 2);
  bf16_t* h1    = (bf16_t*)alloc(BL * 2048 * 2);
  bf16_t* wqc   = (bf16_t*)alloc((size_t)D_ * D_ * 2);
  bf16_t* wkc   = (bf16_t*)alloc((size_t)D_ * D_ * 2);
  bf16_t* wvc   = (bf16_t*)alloc((size_t)D_ * D_ * 2);
  bf16_t* weoc  = (bf16_t*)alloc((size_t)D_ * (H_ * E_) * 2);
  bf16_t* woc   = (bf16_t*)alloc((size_t)D_ * 2 * D_ * 2);
  bf16_t* f1c   = (bf16_t*)alloc((size_t)2048 * D_ * 2);
  bf16_t* f2c   = (bf16_t*)alloc((size_t)D_ * 2048 * 2);

  auto conv = [&](const float* src, bf16_t* dst, int n) {
    f2bf_kernel<<<dim3((n + 255) / 256), dim3(256), 0, stream>>>(src, dst, n);
  };
  conv(wq_W, wqc, D_ * D_);
  conv(wk_W, wkc, D_ * D_);
  conv(wv_W, wvc, D_ * D_);
  conv(weo_W, weoc, D_ * H_ * E_);
  conv(wo_W, woc, D_ * 2 * D_);
  conv(f1_W, f1c, 2048 * D_);
  conv(f2_W, f2c, D_ * 2048);

  // 1) LN(x) -> nx ; edge left/right projections
  ln_kernel<<<dim3((int)BL), dim3(256), 0, stream>>>(x, na_g, na_b, nx, D_);
  lr_kernel<<<dim3((int)BL), dim3(256), 0, stream>>>(x, ttok, lt_W, rt_W, ol_W, ol_b, or_W, lout, rout);

  // 2) QKV projections (WMMA GEMM, bias, bf16 out, (B,L,D) layout)
  dim3 gq((int)(BL / 64), D_ / 64);
  gemm_kernel<<<gq, dim3(128), 0, stream>>>(nx, wqc, wq_b, qb, nullptr, (int)BL, D_, D_, D_, 0, 0);
  gemm_kernel<<<gq, dim3(128), 0, stream>>>(nx, wkc, wk_b, kb, nullptr, (int)BL, D_, D_, D_, 0, 0);
  gemm_kernel<<<gq, dim3(128), 0, stream>>>(nx, wvc, wv_b, vb, nullptr, (int)BL, D_, D_, D_, 0, 0);

  // 3) fused edge -> LN -> kse (bf16) + eb (softcapped)
  edge_kernel<<<dim3((int)BL, L_ / 64), dim3(128), 0, stream>>>(
      lout, rout, wke_W, wke_b, we_W, we_b, ne_g, ne_b, kse, eb);

  // 4) attention: scores+softmax+ctx (ctx written into cat[:, 0:768])
  attn_kernel<<<dim3(B_ * H_, L_ / 16), dim3(128), 0, stream>>>(qb, kb, vb, eb, mask, attn, cat);

  // 5) edge-context stream: ectx = attn @ kse per (b,n), heads on the M axis
  ectx_kernel<<<dim3((int)BL), dim3(128), 0, stream>>>(attn, kse, ectx);

  // 6) ectx @ weo^T + weo_b -> cat[:, 768:1536]
  gemm_kernel<<<gq, dim3(128), 0, stream>>>(ectx, weoc, weo_b, cat, nullptr, (int)BL, D_, H_ * E_, 2 * D_, D_, 0);

  // 7) cat @ wo^T + wo_b + x -> x1 (f32)
  gemm_kernel<<<gq, dim3(128), 0, stream>>>(cat, woc, wo_b, x1, x, (int)BL, D_, 2 * D_, D_, 0, 2);

  // 8) FFN: LN(x1) -> nx2 ; GELU(nx2@f1^T+b) -> h1 ; x1 + h1@f2^T+b -> out
  ln_kernel<<<dim3((int)BL), dim3(256), 0, stream>>>(x1, nf_g, nf_b, nx2, D_);
  gemm_kernel<<<dim3((int)(BL / 64), 2048 / 64), dim3(128), 0, stream>>>(
      nx2, f1c, f1_b, h1, nullptr, (int)BL, 2048, D_, 2048, 0, 1);
  gemm_kernel<<<gq, dim3(128), 0, stream>>>(h1, f2c, f2_b, d_out, x1, (int)BL, D_, 2048, D_, 0, 2);

  (void)in_sizes; (void)n_in; (void)out_size; (void)ws_size;
}
